// TextDecoder_46780783788621
// MI455X (gfx1250) — matrix operand order repaired and checked by reading, not compiled, not run on hardware
//
#include <hip/hip_runtime.h>
#include <hip/hip_bf16.h>
#include <math.h>

typedef __bf16 bf16_t;
typedef __attribute__((ext_vector_type(16))) __bf16 v16bf;
typedef __attribute__((ext_vector_type(8)))  __bf16 v8bf;
typedef __attribute__((ext_vector_type(8)))  float  v8f;

#define BSZ   16
#define SEQL  50
#define NBLK  64
#define DM    256
#define BLKD  768
#define HID   512
#define MEMN  64

// ---- LDS geometry (strides padded: stride % 256B == 16 -> 4-bank shift/row, 16B aligned) ----
#define APAD  776                    // BLKD + 8 bf16 elems  (row stride 1552 B)
#define HPAD  520                    // HID  + 8             (row stride 1040 B)
#define VPAD  264                    // DM   + 8             (row stride  528 B)
#define R0_BYTES   (64 * APAD * 2)   // 99328 : sA  | later sY(f32 64x256)=65536 + sV(bf16 64xVPAD)=33792
#define R1_BYTES   (64 * HPAD * 2)   // 66560 : sH
#define SV_OFF     65536
#define SMALL_OFF  (R0_BYTES + R1_BYTES)     // 165888
#define SMEM_BYTES (SMALL_OFF + 2048)        // 167936 (< 320KB/WG on CDNA5 WGP)

// ---------------- WMMA helpers ----------------
__device__ __forceinline__ v8f wmma_bf16(v16bf a, v16bf b, v8f c) {
  // D = A(16x32 bf16) * B(32x16 bf16) + C(16x16 f32)
  return __builtin_amdgcn_wmma_f32_16x16x32_bf16(false, a, false, b, (short)0, c, false, false);
}

// A fragment from LDS, row-major bf16 with row stride `strideElems`.
// Lane L: row = mBase + (L&15); K chunk = kk*32 + ((L>=16)?8:0) + [0..7] and +16.
__device__ __forceinline__ v16bf load_a_frag(const bf16_t* A, int strideElems, int mBase, int kk) {
  const int lane = threadIdx.x & 31;
  const int row  = mBase + (lane & 15);
  const int kb   = (kk << 5) + ((lane >> 4) << 3);
  const bf16_t* p = A + row * strideElems + kb;
  v8bf lo = *(const v8bf*)(p);         // ds_load_b128
  v8bf hi = *(const v8bf*)(p + 16);    // ds_load_b128
  v16bf r;
#pragma unroll
  for (int i = 0; i < 8; ++i) { r[i] = lo[i]; r[i + 8] = hi[i]; }
  return r;
}

// B fragment from pre-packed weights (see pack_b_kernel): 32B contiguous per lane.
__device__ __forceinline__ v16bf load_b_frag(const bf16_t* __restrict__ P, int kk, int nt, int Nt) {
  const int lane = threadIdx.x & 31;
  const bf16_t* p = P + (((size_t)(kk * Nt + nt) * 32) + lane) * 16;
  return *(const v16bf*)p;
}

// Pack f32 [K][N] row-major -> bf16 fragment-tiled layout matching load_b_frag.
__global__ void pack_b_kernel(const float* __restrict__ W, bf16_t* __restrict__ P, int K, int N) {
  const int Nt = N >> 4;
  const int total = K * N;
  for (int i = blockIdx.x * blockDim.x + threadIdx.x; i < total; i += gridDim.x * blockDim.x) {
    int k = i / N, n = i - k * N;
    int kk = k >> 5, kr = k & 31;
    int nt = n >> 4, nc = n & 15;
    int lane = nc + (((kr >> 3) & 1) << 4);   // lanes 0-15: K 0..7,16..23 ; lanes 16-31: K 8..15,24..31
    int v = (kr & 7) + ((kr >> 4) << 3);
    size_t off = ((size_t)(kk * Nt + nt) * 32 + lane) * 16 + v;
    P[off] = (bf16_t)W[i];
  }
}

// ---------------- core: _blk_trans on 64 gathered rows ----------------
// Result: bf16 normalized rows in sV = sm + SV_OFF, [64][VPAD].
__device__ void blk_trans_64rows(char* sm, const int* sIds,
    const float* __restrict__ emb,
    const bf16_t* __restrict__ pW1, const float* __restrict__ b1,
    const bf16_t* __restrict__ pW2, const float* __restrict__ b2,
    const float* __restrict__ g,   const float* __restrict__ bb)
{
  const int tid  = threadIdx.x;
  const int wave = tid >> 5;
  const int lane = tid & 31;
  bf16_t* sA = (bf16_t*)sm;
  bf16_t* sH = (bf16_t*)(sm + R0_BYTES);
  float*  sY = (float*)sm;                 // reuses sA region after GEMM1
  bf16_t* sV = (bf16_t*)(sm + SV_OFF);     // reuses sA tail after LN

  __syncthreads();                         // cover caller's sIds writes
  // gather embeddings, f32 -> bf16
  for (int i = tid; i < 64 * BLKD; i += 256) {
    int r = i / BLKD, c = i - r * BLKD;
    sA[r * APAD + c] = (bf16_t)emb[(size_t)sIds[r] * BLKD + c];
  }
  __syncthreads();

  // GEMM1: relu(A[64xBLKD] @ W1 + b1) -> sH (bf16)
  for (int t = wave; t < 4 * (HID / 16); t += 8) {
    const int mt = t >> 5, nt = t & 31;
    v8f acc = {};
    for (int kk = 0; kk < BLKD / 32; ++kk) {
      v16bf a = load_a_frag(sA, APAD, mt * 16, kk);
      v16bf b = load_b_frag(pW1, kk, nt, HID / 16);
      acc = wmma_bf16(a, b, acc);
    }
    const int col  = nt * 16 + (lane & 15);
    const int mrow = mt * 16 + ((lane >> 4) << 3);
    const float bias = b1[col];
#pragma unroll
    for (int r = 0; r < 8; ++r) {
      float v = acc[r] + bias;
      sH[(mrow + r) * HPAD + col] = (bf16_t)(v > 0.f ? v : 0.f);
    }
  }
  __syncthreads();

  // GEMM2: sH @ W2 + b2 -> sY (f32)
  for (int t = wave; t < 4 * (DM / 16); t += 8) {
    const int mt = t >> 4, nt = t & 15;
    v8f acc = {};
    for (int kk = 0; kk < HID / 32; ++kk) {
      v16bf a = load_a_frag(sH, HPAD, mt * 16, kk);
      v16bf b = load_b_frag(pW2, kk, nt, DM / 16);
      acc = wmma_bf16(a, b, acc);
    }
    const int col  = nt * 16 + (lane & 15);
    const int mrow = mt * 16 + ((lane >> 4) << 3);
    const float bias = b2[col];
#pragma unroll
    for (int r = 0; r < 8; ++r)
      sY[(mrow + r) * DM + col] = acc[r] + bias;
  }
  __syncthreads();

  // LayerNorm per row -> sV (bf16)
  for (int r = wave; r < 64; r += 8) {
    const float* y = sY + r * DM;
    float s = 0.f;
#pragma unroll
    for (int j = 0; j < 8; ++j) s += y[lane + 32 * j];
#pragma unroll
    for (int m = 16; m; m >>= 1) s += __shfl_xor(s, m, 32);
    const float mu = s * (1.f / DM);
    float v2 = 0.f;
#pragma unroll
    for (int j = 0; j < 8; ++j) { float d = y[lane + 32 * j] - mu; v2 += d * d; }
#pragma unroll
    for (int m = 16; m; m >>= 1) v2 += __shfl_xor(v2, m, 32);
    const float rstd = __frsqrt_rn(v2 * (1.f / DM) + 1e-5f);
#pragma unroll
    for (int j = 0; j < 8; ++j) {
      int c = lane + 32 * j;
      sV[r * VPAD + c] = (bf16_t)(g[c] * (y[c] - mu) * rstd + bb[c]);
    }
  }
  __syncthreads();
}

// ---------------- phase 1: per-(b,l) block selection ----------------
__global__ void __launch_bounds__(256) select_blocks_kernel(
    const float* __restrict__ tva, const float* __restrict__ emb,
    const bf16_t* __restrict__ pW1, const float* __restrict__ b1,
    const bf16_t* __restrict__ pW2, const float* __restrict__ b2,
    const float* __restrict__ g, const float* __restrict__ bb,
    const float* __restrict__ am_w1, const float* __restrict__ am_b1,
    const float* __restrict__ am_w2, const bf16_t* __restrict__ pAw1h,
    const int* __restrict__ input_txt, const int* __restrict__ masks_txt,
    int* __restrict__ sel_ids, int* __restrict__ sel_mask)
{
  extern __shared__ char sm[];
  const int tid = threadIdx.x, wave = tid >> 5, lane = tid & 31;
  const int bl = blockIdx.x;                 // b*SEQL + l
  int*   sIds   = (int*)(sm + SMALL_OFF);
  int*   sMask  = sIds + 64;
  float* sC     = (float*)(sMask + 64);      // 256 f32
  float* sScore = sC + 256;                  // 64
  float* sE     = sScore + 64;               // 64 (scratch)
  float* sS     = (float*)sm;                // over sY region (free after LN)
  bf16_t* sV    = (bf16_t*)(sm + SV_OFF);

  if (tid < 64) {
    sIds[tid]  = input_txt[(size_t)bl * NBLK + tid];
    sMask[tid] = masks_txt[(size_t)bl * NBLK + tid];
  }
  { // c[j] = am_b1[j] + sum_d tv[d]*am_w1[d][j]  (tv part identical for all NB rows)
    const float* tv = tva + (size_t)bl * DM;
    float c = am_b1[tid];
    for (int d = 0; d < DM; ++d) c += tv[d] * am_w1[d * DM + tid];
    sC[tid] = c;
  }
  blk_trans_64rows(sm, sIds, emb, pW1, b1, pW2, b2, g, bb);

  // GEMM3: S = v_text @ am_w1[DM:2DM,:]  -> sS (f32)
  for (int t = wave; t < 64; t += 8) {
    const int mt = t >> 4, nt = t & 15;
    v8f acc = {};
    for (int kk = 0; kk < DM / 32; ++kk) {
      v16bf a  = load_a_frag(sV, VPAD, mt * 16, kk);
      v16bf bf = load_b_frag(pAw1h, kk, nt, DM / 16);
      acc = wmma_bf16(a, bf, acc);
    }
    const int col  = nt * 16 + (lane & 15);
    const int mrow = mt * 16 + ((lane >> 4) << 3);
#pragma unroll
    for (int r = 0; r < 8; ++r) sS[(mrow + r) * DM + col] = acc[r];
  }
  __syncthreads();

  // score_r = sum_j tanh(S + c)_j * am_w2_j
  for (int r = wave; r < 64; r += 8) {
    float s = 0.f;
#pragma unroll
    for (int j = 0; j < 8; ++j) {
      int c = lane + 32 * j;
      s += tanhf(sS[r * DM + c] + sC[c]) * am_w2[c];
    }
#pragma unroll
    for (int m = 16; m; m >>= 1) s += __shfl_xor(s, m, 32);
    if (lane == 0) sScore[r] = s;
  }
  __syncthreads();

  // softmax over 64, mask->0, exact top_k(64 of 64) via rank (lax.top_k tie-break)
  if (tid == 0) { float mx = sScore[0]; for (int j = 1; j < 64; ++j) mx = fmaxf(mx, sScore[j]); sC[0] = mx; }
  __syncthreads();
  float ex = 0.f;
  if (tid < 64) { ex = __expf(sScore[tid] - sC[0]); sE[tid] = ex; }
  __syncthreads();
  if (tid == 0) { float s = 0.f; for (int j = 0; j < 64; ++j) s += sE[j]; sC[1] = s; }
  __syncthreads();
  float p = 0.f;
  if (tid < 64) { p = (sMask[tid] != 0) ? ex / sC[1] : 0.f; sE[tid] = p; }
  __syncthreads();
  if (tid < 64) {
    int rank = 0;
    for (int j = 0; j < 64; ++j) {
      float pj = sE[j];
      rank += (pj > p) || (pj == p && j < tid);
    }
    size_t base = (size_t)bl * NBLK;
    sel_ids[base + rank]  = sIds[tid];
    sel_mask[base + rank] = sMask[tid];
  }
}

// ---------------- QW[b,l,:] = (tva+v_all)[b,l,:] @ wq ----------------
__global__ void qw_kernel(const float* __restrict__ v_all, const float* __restrict__ tva,
                          const float* __restrict__ wq, float* __restrict__ QW)
{
  __shared__ float q[DM];
  const int bl = blockIdx.x, tid = threadIdx.x;
  q[tid] = tva[(size_t)bl * DM + tid] + v_all[(size_t)bl * DM + tid];
  __syncthreads();
  float s = 0.f;
  for (int d = 0; d < DM; ++d) s += q[d] * wq[d * DM + tid];
  QW[(size_t)bl * DM + tid] = s;
}

__global__ void init_carry_kernel(const int* __restrict__ sel_ids, const int* __restrict__ sel_mask,
                                  int* __restrict__ cid, int* __restrict__ cmk)
{
  const int b = blockIdx.x, m = threadIdx.x;
  cid[b * MEMN + m] = sel_ids[(size_t)(b * SEQL) * NBLK + m];
  cmk[b * MEMN + m] = sel_mask[(size_t)(b * SEQL) * NBLK + m];
}

// ---------------- per-step: e[b,m] = (q wq) . (blk_trans(mem_in) wk) / 16 ----------------
__global__ void __launch_bounds__(256) mem_x_kernel(
    const float* __restrict__ emb,
    const bf16_t* __restrict__ pW1, const float* __restrict__ b1,
    const bf16_t* __restrict__ pW2, const float* __restrict__ b2,
    const float* __restrict__ g, const float* __restrict__ bb,
    const bf16_t* __restrict__ pWk, const float* __restrict__ QW, int t,
    const int* __restrict__ cur_ids, const int* __restrict__ sel_ids,
    float* __restrict__ E)
{
  extern __shared__ char sm[];
  const int tid = threadIdx.x, wave = tid >> 5, lane = tid & 31;
  const int b = blockIdx.x >> 1, half = blockIdx.x & 1;
  int*   sIds = (int*)(sm + SMALL_OFF);
  float* sE   = (float*)(sIds + 64 + 64 + 256 + 64);   // same slot as kernel1's sE
  bf16_t* sV  = (bf16_t*)(sm + SV_OFF);

  if (tid < 64) {
    sIds[tid] = half ? sel_ids[((size_t)(b * SEQL + t)) * NBLK + tid]
                     : cur_ids[b * MEMN + tid];
    sE[tid] = 0.f;
  }
  blk_trans_64rows(sm, sIds, emb, pW1, b1, pW2, b2, g, bb);

  const float* qw = QW + (size_t)(b * SEQL + t) * DM;
  for (int tt = wave; tt < 64; tt += 8) {
    const int mt = tt >> 4, nt = tt & 15;
    v8f acc = {};
    for (int kk = 0; kk < DM / 32; ++kk) {
      v16bf a  = load_a_frag(sV, VPAD, mt * 16, kk);
      v16bf bf = load_b_frag(pWk, kk, nt, DM / 16);
      acc = wmma_bf16(a, bf, acc);
    }
    const float qv = qw[nt * 16 + (lane & 15)];
    const int mrow = mt * 16 + ((lane >> 4) << 3);
#pragma unroll
    for (int r = 0; r < 8; ++r) {
      float p = acc[r] * qv;                 // one column's contribution to row mrow+r
      p += __shfl_xor(p, 1, 32);
      p += __shfl_xor(p, 2, 32);
      p += __shfl_xor(p, 4, 32);
      p += __shfl_xor(p, 8, 32);             // reduce the 16 cols of this tile
      if ((lane & 15) == 0) atomicAdd(&sE[mrow + r], p);   // ds_add_f32
    }
  }
  __syncthreads();
  if (tid < 64) E[(size_t)b * 128 + half * 64 + tid] = sE[tid] * 0.0625f;  // / sqrt(256)
}

// ---------------- per-step: softmax+mask+top_k(64 of 128), write carry & stack ----------------
__global__ void mem_select_kernel(
    const float* __restrict__ E,
    const int* __restrict__ cur_ids, const int* __restrict__ cur_mk,
    const int* __restrict__ sel_ids, const int* __restrict__ sel_mk, int t,
    int* __restrict__ nxt_ids, int* __restrict__ nxt_mk, int* __restrict__ mem_stack)
{
  __shared__ float sp[128];
  __shared__ int sid[128], smk[128];
  __shared__ float red[2];
  const int b = blockIdx.x, tid = threadIdx.x;
  float e = E[(size_t)b * 128 + tid];
  int id, mk;
  if (tid < 64) { id = cur_ids[b * MEMN + tid]; mk = cur_mk[b * MEMN + tid]; }
  else {
    size_t base = ((size_t)(b * SEQL + t)) * NBLK + (tid - 64);
    id = sel_ids[base]; mk = sel_mk[base];
  }
  sid[tid] = id; smk[tid] = mk; sp[tid] = e;
  __syncthreads();
  if (tid == 0) { float mx = sp[0]; for (int j = 1; j < 128; ++j) mx = fmaxf(mx, sp[j]); red[0] = mx; }
  __syncthreads();
  float ex = __expf(e - red[0]);
  sp[tid] = ex;
  __syncthreads();
  if (tid == 0) { float s = 0.f; for (int j = 0; j < 128; ++j) s += sp[j]; red[1] = s; }
  __syncthreads();
  float p = mk ? ex / red[1] : 0.f;
  sp[tid] = p;
  __syncthreads();
  int rank = 0;
  for (int j = 0; j < 128; ++j) {
    float pj = sp[j];
    rank += (pj > p) || (pj == p && j < tid);
  }
  if (rank < MEMN) {
    nxt_ids[b * MEMN + rank] = id;
    nxt_mk[b * MEMN + rank]  = mk;
    mem_stack[((size_t)b * (SEQL - 1) + (t - 1)) * MEMN + rank] = id;
  }
  (void)sid; (void)smk;
}

// ---------------- final: memory_final gather, blk_trans+maxpool, masked maxpool ----------------
__global__ void __launch_bounds__(256) final_pool_kernel(
    const float* __restrict__ emb,
    const bf16_t* __restrict__ pW1, const float* __restrict__ b1,
    const bf16_t* __restrict__ pW2, const float* __restrict__ b2,
    const float* __restrict__ g, const float* __restrict__ bb,
    const int* __restrict__ mem_stack, const int* __restrict__ lengths,
    const float* __restrict__ tva,
    int* __restrict__ mfin, float* __restrict__ vfin, float* __restrict__ memvec)
{
  extern __shared__ char sm[];
  const int tid = threadIdx.x, b = blockIdx.x;
  int* sIds = (int*)(sm + SMALL_OFF);
  bf16_t* sV = (bf16_t*)(sm + SV_OFF);
  if (tid < 64) {
    int li = lengths[b] - 2;
    int id = mem_stack[((size_t)b * (SEQL - 1) + li) * MEMN + tid];
    sIds[tid] = id;
    mfin[b * MEMN + tid] = id;
  }
  blk_trans_64rows(sm, sIds, emb, pW1, b1, pW2, b2, g, bb);
  if (tid < DM) {
    float m = -INFINITY;
    for (int r = 0; r < 64; ++r) m = fmaxf(m, (float)sV[r * VPAD + tid]);
    memvec[b * DM + tid] = m;
    float vm = -INFINITY;
    const int len = lengths[b];
    for (int l2 = 0; l2 < len; ++l2) vm = fmaxf(vm, tva[((size_t)b * SEQL + l2) * DM + tid]);
    vfin[b * DM + tid] = vm;
  }
}

__global__ void head_kernel(const float* __restrict__ vfin, const float* __restrict__ memvec,
                            const float* __restrict__ out_w, const float* __restrict__ out_b,
                            const int* __restrict__ mfin, float* __restrict__ out)
{
  const int tid = threadIdx.x;
  if (tid < 32) {
    const int b = tid >> 1, k = tid & 1;
    float s = out_b[k];
    for (int j = 0; j < DM; ++j) s += vfin[b * DM + j] * out_w[j * 2 + k];
    for (int j = 0; j < DM; ++j) s += memvec[b * DM + j] * out_w[(DM + j) * 2 + k];
    out[b * 2 + k] = s;
  }
  for (int i = tid; i < BSZ * MEMN; i += 256) out[32 + i] = (float)mfin[i];
}

// ---------------- host ----------------
extern "C" void kernel_launch(void* const* d_in, const int* in_sizes, int n_in,
                              void* d_out, int out_size, void* d_ws, size_t ws_size,
                              hipStream_t stream) {
  const float* v_all = (const float*)d_in[0];
  const float* tva   = (const float*)d_in[1];
  const float* emb   = (const float*)d_in[2];
  const float* w1    = (const float*)d_in[3];
  const float* b1    = (const float*)d_in[4];
  const float* w2    = (const float*)d_in[5];
  const float* b2    = (const float*)d_in[6];
  const float* g     = (const float*)d_in[7];
  const float* bb    = (const float*)d_in[8];
  const float* wq    = (const float*)d_in[9];
  const float* wk    = (const float*)d_in[10];
  const float* am_w1 = (const float*)d_in[11];
  const float* am_b1 = (const float*)d_in[12];
  const float* am_w2 = (const float*)d_in[13];
  const float* out_w = (const float*)d_in[14];
  const float* out_b = (const float*)d_in[15];
  const int* input_txt = (const int*)d_in[16];
  const int* masks_txt = (const int*)d_in[17];
  const int* lengths   = (const int*)d_in[18];
  float* out = (float*)d_out;

  char* ws = (char*)d_ws;
  size_t o = 0;
  auto take = [&](size_t bytes) { char* p = ws + o; o += (bytes + 255) & ~(size_t)255; return p; };
  bf16_t* pW1    = (bf16_t*)take((size_t)BLKD * HID * 2);
  bf16_t* pW2    = (bf16_t*)take((size_t)HID * DM * 2);
  bf16_t* pAw1h  = (bf16_t*)take((size_t)DM * DM * 2);
  bf16_t* pWk    = (bf16_t*)take((size_t)DM * DM * 2);
  float*  QW     = (float*)take((size_t)BSZ * SEQL * DM * 4);
  int* sel_ids   = (int*)take((size_t)BSZ * SEQL * NBLK * 4);
  int* sel_mask  = (int*)take((size_t)BSZ * SEQL * NBLK * 4);
  int* carA_id   = (int*)take(BSZ * MEMN * 4);
  int* carA_mk   = (int*)take(BSZ * MEMN * 4);
  int* carB_id   = (int*)take(BSZ * MEMN * 4);
  int* carB_mk   = (int*)take(BSZ * MEMN * 4);
  float* E       = (float*)take(BSZ * 128 * 4);
  int* mem_stack = (int*)take((size_t)BSZ * (SEQL - 1) * MEMN * 4);
  float* vfin    = (float*)take(BSZ * DM * 4);
  float* memvec  = (float*)take(BSZ * DM * 4);
  int* mfin      = (int*)take(BSZ * MEMN * 4);

  hipFuncSetAttribute((const void*)select_blocks_kernel,
                      hipFuncAttributeMaxDynamicSharedMemorySize, SMEM_BYTES);
  hipFuncSetAttribute((const void*)mem_x_kernel,
                      hipFuncAttributeMaxDynamicSharedMemorySize, SMEM_BYTES);
  hipFuncSetAttribute((const void*)final_pool_kernel,
                      hipFuncAttributeMaxDynamicSharedMemorySize, SMEM_BYTES);

  // weight packing (bf16, WMMA-B fragment layout)
  pack_b_kernel<<<512, 256, 0, stream>>>(w1, pW1, BLKD, HID);
  pack_b_kernel<<<256, 256, 0, stream>>>(w2, pW2, HID, DM);
  pack_b_kernel<<<128, 256, 0, stream>>>(am_w1 + DM * DM, pAw1h, DM, DM);
  pack_b_kernel<<<128, 256, 0, stream>>>(wk, pWk, DM, DM);

  qw_kernel<<<BSZ * SEQL, 256, 0, stream>>>(v_all, tva, wq, QW);

  select_blocks_kernel<<<BSZ * SEQL, 256, SMEM_BYTES, stream>>>(
      tva, emb, pW1, b1, pW2, b2, g, bb, am_w1, am_b1, am_w2, pAw1h,
      input_txt, masks_txt, sel_ids, sel_mask);

  init_carry_kernel<<<BSZ, 64, 0, stream>>>(sel_ids, sel_mask, carA_id, carA_mk);

  int* cur_id = carA_id; int* cur_mk = carA_mk;
  int* nxt_id = carB_id; int* nxt_mk = carB_mk;
  for (int t = 1; t < SEQL; ++t) {
    mem_x_kernel<<<BSZ * 2, 256, SMEM_BYTES, stream>>>(
        emb, pW1, b1, pW2, b2, g, bb, pWk, QW, t, cur_id, sel_ids, E);
    mem_select_kernel<<<BSZ, 128, 0, stream>>>(
        E, cur_id, cur_mk, sel_ids, sel_mask, t, nxt_id, nxt_mk, mem_stack);
    int* ti = cur_id; cur_id = nxt_id; nxt_id = ti;
    int* tm = cur_mk; cur_mk = nxt_mk; nxt_mk = tm;
  }

  final_pool_kernel<<<BSZ, 256, SMEM_BYTES, stream>>>(
      emb, pW1, b1, pW2, b2, g, bb, mem_stack, lengths, tva, mfin, vfin, memvec);
  head_kernel<<<1, 256, 0, stream>>>(vfin, memvec, out_w, out_b, mfin, out);

  (void)in_sizes; (void)n_in; (void)out_size; (void)ws_size;
}
